// StrokeGroupedAttention_50379966382568
// MI455X (gfx1250) — matrix operand order repaired and checked by reading, not compiled, not run on hardware
//
#include <hip/hip_runtime.h>
#include <hip/hip_bf16.h>

typedef __attribute__((ext_vector_type(16))) _Float16 v16h;
typedef __attribute__((ext_vector_type(8)))  _Float16 v8h;
typedef __attribute__((ext_vector_type(8)))  float    v8f;
typedef __attribute__((ext_vector_type(4)))  float    v4f;

#define E_DIM  1024
#define HEADS  16
#define HD     64

// ---- vectorized 8-element fragment loaders (16B-aligned sources) ----------
__device__ __forceinline__ void load8(v16h& d, int off, const _Float16* p)
{
    const v8h t = *(const v8h*)p;
#pragma unroll
    for (int i = 0; i < 8; ++i) d[off + i] = t[i];
}
__device__ __forceinline__ void load8(v16h& d, int off, const float* p)
{
    const v4f a = *(const v4f*)p;
    const v4f b = *(const v4f*)(p + 4);
#pragma unroll
    for (int i = 0; i < 4; ++i) {
        d[off + i]     = (_Float16)a[i];
        d[off + 4 + i] = (_Float16)b[i];
    }
}

// ---------------------------------------------------------------------------
// f32 -> f16 elementwise conversion (weights), 8 elements per thread.
// ---------------------------------------------------------------------------
__global__ __launch_bounds__(256) void cvt_f32_f16(const float* __restrict__ src,
                                                   _Float16* __restrict__ dst,
                                                   int n8)
{
    const int i = blockIdx.x * blockDim.x + threadIdx.x;
    if (i >= n8) return;
    const unsigned o = (unsigned)i * 8u;
    const v4f a = *(const v4f*)(src + o);
    const v4f b = *(const v4f*)(src + o + 4);
    v8h d;
#pragma unroll
    for (int e = 0; e < 4; ++e) { d[e] = (_Float16)a[e]; d[4 + e] = (_Float16)b[e]; }
    *(v8h*)(dst + o) = d;
}

// ---------------------------------------------------------------------------
// C = A[M,K] * W[N,K]^T + bias[N].  One wave -> 32x64 strip (2 m-tiles x 4
// n-tiles; B fragments shared across both m-tiles).  N % 64 == 0.  W is f16.
// TRANS=false: C[m*ldC + n];  TRANS=true: C[n*ldC + m] (b128 stores).
// ---------------------------------------------------------------------------
template <typename AT, typename OT, bool TRANS>
__global__ __launch_bounds__(256) void gemm_bias_wmma(
    const AT* __restrict__ A, const _Float16* __restrict__ W,
    const float* __restrict__ bias, OT* __restrict__ C,
    int M, int N, int K, int ldC)
{
    const int lane   = threadIdx.x & 31;
    const int lane15 = lane & 15;
    const int half   = lane >> 4;
    const int mtiles = (M + 31) >> 5;          // 32-row strips
    const int ngrp   = N >> 6;
    const int wid    = __builtin_amdgcn_readfirstlane(
        blockIdx.x * (blockDim.x >> 5) + (threadIdx.x >> 5));
    if (wid >= mtiles * ngrp) return;
    const int mt = wid / ngrp;
    const int ng = wid % ngrp;

    int arow0 = mt * 32 + lane15;       if (arow0 >= M) arow0 = M - 1;
    int arow1 = mt * 32 + 16 + lane15;  if (arow1 >= M) arow1 = M - 1;

    v8f acc[2][4] = { { {}, {}, {}, {} }, { {}, {}, {}, {} } };
    for (int kc = 0; kc < K; kc += 32) {
        v16h a0, a1;
        const AT* ap0 = A + (unsigned)arow0 * (unsigned)K + (unsigned)(kc + half * 8);
        const AT* ap1 = A + (unsigned)arow1 * (unsigned)K + (unsigned)(kc + half * 8);
        load8(a0, 0, ap0);  load8(a0, 8, ap0 + 16);
        load8(a1, 0, ap1);  load8(a1, 8, ap1 + 16);
#pragma unroll
        for (int t = 0; t < 4; ++t) {
            const unsigned wr = (unsigned)(ng * 64 + t * 16 + lane15);
            const _Float16* wp = W + wr * (unsigned)K + (unsigned)(kc + half * 8);
            v16h b;
            load8(b, 0, wp);
            load8(b, 8, wp + 16);
            acc[0][t] = __builtin_amdgcn_wmma_f32_16x16x32_f16(
                false, a0, false, b, (short)0, acc[0][t], false, false);
            acc[1][t] = __builtin_amdgcn_wmma_f32_16x16x32_f16(
                false, a1, false, b, (short)0, acc[1][t], false, false);
        }
    }

#pragma unroll
    for (int mi = 0; mi < 2; ++mi) {
#pragma unroll
        for (int t = 0; t < 4; ++t) {
            const int n = ng * 64 + t * 16 + lane15;
            const float bv = bias[n];
            if constexpr (TRANS) {
                const int m0 = mt * 32 + mi * 16 + half * 8;
                v8h outv;
#pragma unroll
                for (int e = 0; e < 8; ++e) outv[e] = (_Float16)(acc[mi][t][e] + bv);
                if (m0 + 8 <= M) {
                    *(v8h*)(C + (unsigned)n * (unsigned)ldC + (unsigned)m0) = outv;
                } else {
#pragma unroll
                    for (int e = 0; e < 8; ++e)
                        if (m0 + e < M)
                            C[(unsigned)n * (unsigned)ldC + (unsigned)(m0 + e)] = outv[e];
                }
            } else {
#pragma unroll
                for (int e = 0; e < 8; ++e) {
                    const int row = mt * 32 + mi * 16 + e + half * 8;
                    if (row < M)
                        C[(unsigned)row * (unsigned)ldC + (unsigned)n] =
                            (OT)(acc[mi][t][e] + bv);
                }
            }
        }
    }
}

// ---------------------------------------------------------------------------
// Wave-level flash attention.  K row-major [tok][E]; V transposed [E][tok].
// All control values are SGPR-uniform (readfirstlane) so WMMA sees EXEC=~0.
// ---------------------------------------------------------------------------
struct AttnParams {
    const _Float16* q; const _Float16* k; const _Float16* vt;
    _Float16* o;
    int qs, ks, vt_ld, os;
    int Lq, Lk, ngroups, causal, clsmode;
};

__global__ __launch_bounds__(256) void flash_attn_wmma(AttnParams p)
{
    const int lane   = threadIdx.x & 31;
    const int lane15 = lane & 15;
    const int lw     = threadIdx.x >> 5;
    const int half   = lane >> 4;
    const int qtiles = (p.Lq + 15) >> 4;
    const int wid    = __builtin_amdgcn_readfirstlane(
        blockIdx.x * (blockDim.x >> 5) + lw);
    if (wid >= p.ngroups * HEADS * qtiles) return;
    const int qt = wid % qtiles;
    const int h  = (wid / qtiles) % HEADS;
    const int g  = wid / (qtiles * HEADS);

    __shared__ _Float16 lds[8][16][32];

    // --- Q fragments, pre-scaled by 1/sqrt(hd)=0.125 (exact in f16) ---
    v16h aq0, aq1;
    {
        int rr = qt * 16 + lane15;
        if (rr >= p.Lq) rr = p.Lq - 1;
        const unsigned qrow = p.clsmode ? (unsigned)((g & 3) * 16 + (g >> 2))
                                        : (unsigned)(g * p.Lq + rr);
        const _Float16* qp = p.q + qrow * (unsigned)p.qs
                                 + (unsigned)(h * HD + half * 8);
        load8(aq0, 0, qp);       load8(aq0, 8, qp + 16);
        load8(aq1, 0, qp + 32);  load8(aq1, 8, qp + 48);
        const _Float16 sc = (_Float16)0.125f;
#pragma unroll
        for (int i = 0; i < 16; ++i) { aq0[i] *= sc; aq1[i] *= sc; }
    }

    float m_r[8], l_r[8];
    v8f accO[4] = { {}, {}, {}, {} };
#pragma unroll
    for (int e = 0; e < 8; ++e) { m_r[e] = -1e30f; l_r[e] = 0.0f; }

    const int kend = p.causal ? ((qt * 16 + 16) < p.Lk ? (qt * 16 + 16) : p.Lk)
                              : p.Lk;
    const unsigned kgbase = (unsigned)(g * p.Lk);

    for (int kb = 0; kb < kend; kb += 32) {
        // --- scores ---
        v8f s0 = {}, s1 = {};
#pragma unroll
        for (int j = 0; j < 2; ++j) {
            v8f s = {};
            const int key  = kb + j * 16 + lane15;
            const int keyc = key < p.Lk ? key : p.Lk - 1;
            const _Float16* kp = p.k + (kgbase + (unsigned)keyc) * (unsigned)p.ks
                                     + (unsigned)(h * HD + half * 8);
            v16h bk;
            load8(bk, 0, kp);       load8(bk, 8, kp + 16);
            s = __builtin_amdgcn_wmma_f32_16x16x32_f16(false, aq0, false, bk,
                                                       (short)0, s, false, false);
            load8(bk, 0, kp + 32);  load8(bk, 8, kp + 48);
            s = __builtin_amdgcn_wmma_f32_16x16x32_f16(false, aq1, false, bk,
                                                       (short)0, s, false, false);
            // branchless masking -> v_cndmask, no exec manipulation
#pragma unroll
            for (int e = 0; e < 8; ++e) {
                const int qrow = qt * 16 + e + half * 8;
                const int dead = (int)(key >= p.Lk) | (p.causal & (int)(key > qrow));
                s[e] = dead ? -1e30f : s[e];
            }
            if (j == 0) s0 = s; else s1 = s;
        }

        // --- online softmax (row reductions within 16-lane halves) ---
#pragma unroll
        for (int e = 0; e < 8; ++e) {
            float t = fmaxf(s0[e], s1[e]);
#pragma unroll
            for (int m = 1; m < 16; m <<= 1) t = fmaxf(t, __shfl_xor(t, m, 32));
            const float mnew = fmaxf(m_r[e], t);
            const float sc   = __expf(m_r[e] - mnew);
            const float p0   = __expf(s0[e] - mnew);
            const float p1   = __expf(s1[e] - mnew);
            float rs = p0 + p1;
#pragma unroll
            for (int m = 1; m < 16; m <<= 1) rs += __shfl_xor(rs, m, 32);
            l_r[e] = l_r[e] * sc + rs;
            m_r[e] = mnew;
#pragma unroll
            for (int t4 = 0; t4 < 4; ++t4) accO[t4][e] *= sc;
            s0[e] = p0; s1[e] = p1;
        }

        // --- P: C-layout -> A-layout via per-wave LDS slot ---
#pragma unroll
        for (int e = 0; e < 8; ++e) {
            lds[lw][half * 8 + e][lane15]      = (_Float16)s0[e];
            lds[lw][half * 8 + e][16 + lane15] = (_Float16)s1[e];
        }
        asm volatile("s_wait_dscnt 0x0" ::: "memory");
        v16h pa;
        {
            const _Float16* lp = &lds[lw][lane15][half * 8];
            load8(pa, 0, lp);
            load8(pa, 8, lp + 16);
        }
        asm volatile("s_wait_dscnt 0x0" ::: "memory");

        // --- O += P(16x32) * V(32x64): V^T rows contiguous in keys ---
        const unsigned vcol = kgbase + (unsigned)(kb + half * 8);
        if (kb + 32 <= p.Lk) {          // wave-uniform fast path
#pragma unroll
            for (int t4 = 0; t4 < 4; ++t4) {
                const _Float16* vp = p.vt +
                    (unsigned)(h * HD + t4 * 16 + lane15) * (unsigned)p.vt_ld + vcol;
                v16h bv;
                load8(bv, 0, vp);
                load8(bv, 8, vp + 16);
                accO[t4] = __builtin_amdgcn_wmma_f32_16x16x32_f16(
                    false, pa, false, bv, (short)0, accO[t4], false, false);
            }
        } else {                         // uniform tail: zero-fill invalid keys
#pragma unroll
            for (int t4 = 0; t4 < 4; ++t4) {
                const _Float16* vp = p.vt +
                    (unsigned)(h * HD + t4 * 16 + lane15) * (unsigned)p.vt_ld + vcol;
                v16h bv;
#pragma unroll
                for (int i = 0; i < 16; ++i) {
                    const int o = i + (i & 8);
                    const int key = kb + half * 8 + o;
                    bv[i] = (key < p.Lk) ? vp[o] : (_Float16)0.0f;
                }
                accO[t4] = __builtin_amdgcn_wmma_f32_16x16x32_f16(
                    false, pa, false, bv, (short)0, accO[t4], false, false);
            }
        }
    }

    // --- normalize + write f16 output ---
#pragma unroll
    for (int e = 0; e < 8; ++e) {
        const int rr = qt * 16 + e + half * 8;
        if (rr >= p.Lq) continue;
        const unsigned orow = p.clsmode ? (unsigned)((g & 3) * 16 + (g >> 2))
                                        : (unsigned)(g * p.Lq + rr);
        const float inv = 1.0f / fmaxf(l_r[e], 1e-20f);
#pragma unroll
        for (int t4 = 0; t4 < 4; ++t4)
            p.o[orow * (unsigned)p.os + (unsigned)(h * HD + t4 * 16 + lane15)] =
                (_Float16)(accO[t4][e] * inv);
    }
}

// ---------------------------------------------------------------------------
// f16 context [4][1040][E] = per-batch concat(text_out, cls_out).
// ---------------------------------------------------------------------------
__global__ __launch_bounds__(256) void gather_ctx(const float* __restrict__ text_out,
                                                  const float* __restrict__ cls_out,
                                                  _Float16* __restrict__ ctx)
{
    const int idx = blockIdx.x * blockDim.x + threadIdx.x;
    const int total = 4 * 1040 * (E_DIM / 8);
    if (idx >= total) return;
    const int row  = idx >> 7;
    const int col0 = (idx & 127) * 8;
    const int b = row / 1040;
    const int t = row % 1040;
    const float* src = (t < 1024)
        ? text_out + (unsigned)(b * 1024 + t) * E_DIM + col0
        : cls_out  + (unsigned)(b * 16 + (t - 1024)) * E_DIM + col0;
    const v4f a = *(const v4f*)src;
    const v4f c = *(const v4f*)(src + 4);
    v8h o;
#pragma unroll
    for (int i = 0; i < 4; ++i) { o[i] = (_Float16)a[i]; o[4 + i] = (_Float16)c[i]; }
    *(v8h*)(ctx + (unsigned)row * E_DIM + col0) = o;
}

// ---------------------------------------------------------------------------
static inline int gemm_blocks(int M, int N)
{
    const long long waves = (long long)((M + 31) / 32) * (N / 64);
    return (int)((waves + 7) / 8);
}

extern "C" void kernel_launch(void* const* d_in, const int* in_sizes, int n_in,
                              void* d_out, int out_size, void* d_ws, size_t ws_size,
                              hipStream_t stream)
{
    (void)in_sizes; (void)n_in; (void)out_size; (void)ws_size;

    const float* text_tokens  = (const float*)d_in[0];
    const float* patch_groups = (const float*)d_in[1];
    const float* cls_tokens   = (const float*)d_in[2];
    const float* int_token    = (const float*)d_in[3];
    const float* text_w_in    = (const float*)d_in[4];
    const float* text_b_in    = (const float*)d_in[5];
    const float* text_w_out   = (const float*)d_in[6];
    const float* text_b_out   = (const float*)d_in[7];
    const float* patch_w_in   = (const float*)d_in[8];
    const float* patch_b_in   = (const float*)d_in[9];
    const float* patch_w_out  = (const float*)d_in[10];
    const float* patch_b_out  = (const float*)d_in[11];
    const float* int_w_in     = (const float*)d_in[12];
    const float* int_b_in     = (const float*)d_in[13];
    const float* int_w_out    = (const float*)d_in[14];
    const float* int_b_out    = (const float*)d_in[15];

    float* out_text  = (float*)d_out;                    // [4096,1024]
    float* out_patch = out_text  + (size_t)4096 * 1024;  // [16384,1024]
    float* out_cls   = out_patch + (size_t)16384 * 1024; // [64,1024]
    float* out_int   = out_cls   + (size_t)64 * 1024;    // [4,1024]

    char* ws = (char*)d_ws;
    _Float16* text_qk    = (_Float16*)ws; ws += (size_t)4096  * 2048 * 2;
    _Float16* text_vT    = (_Float16*)ws; ws += (size_t)1024  * 4096 * 2;
    _Float16* patch_qk   = (_Float16*)ws; ws += (size_t)16384 * 2048 * 2;
    _Float16* patch_vT   = (_Float16*)ws; ws += (size_t)1024  * 16384 * 2;
    _Float16* cls_q      = (_Float16*)ws; ws += (size_t)64    * 1024 * 2;
    _Float16* int_q      = (_Float16*)ws; ws += (size_t)4     * 1024 * 2;
    _Float16* text_attn  = (_Float16*)ws; ws += (size_t)4096  * 1024 * 2;
    _Float16* patch_attn = (_Float16*)ws; ws += (size_t)16384 * 1024 * 2;
    _Float16* cls_attn   = (_Float16*)ws; ws += (size_t)64    * 1024 * 2;
    _Float16* ctx        = (_Float16*)ws; ws += (size_t)4160  * 1024 * 2;
    _Float16* ctx_k      = (_Float16*)ws; ws += (size_t)4160  * 1024 * 2;
    _Float16* ctx_vT     = (_Float16*)ws; ws += (size_t)1024  * 4160 * 2;
    _Float16* int_attn   = (_Float16*)ws; ws += (size_t)4     * 1024 * 2;
    _Float16* w_text_in  = (_Float16*)ws; ws += (size_t)3072  * 1024 * 2;
    _Float16* w_patch_in = (_Float16*)ws; ws += (size_t)3072  * 1024 * 2;
    _Float16* w_int_in   = (_Float16*)ws; ws += (size_t)3072  * 1024 * 2;
    _Float16* w_text_o   = (_Float16*)ws; ws += (size_t)1024  * 1024 * 2;
    _Float16* w_patch_o  = (_Float16*)ws; ws += (size_t)1024  * 1024 * 2;
    _Float16* w_int_o    = (_Float16*)ws; ws += (size_t)1024  * 1024 * 2;

    const dim3 blk(256);

    // ---- one-shot weight conversion f32 -> f16 ----
    const int n8_in  = 3072 * 1024 / 8, n8_o = 1024 * 1024 / 8;
    cvt_f32_f16<<<(n8_in + 255) / 256, blk, 0, stream>>>(text_w_in,  w_text_in,  n8_in);
    cvt_f32_f16<<<(n8_in + 255) / 256, blk, 0, stream>>>(patch_w_in, w_patch_in, n8_in);
    cvt_f32_f16<<<(n8_in + 255) / 256, blk, 0, stream>>>(int_w_in,   w_int_in,   n8_in);
    cvt_f32_f16<<<(n8_o  + 255) / 256, blk, 0, stream>>>(text_w_out,  w_text_o,  n8_o);
    cvt_f32_f16<<<(n8_o  + 255) / 256, blk, 0, stream>>>(patch_w_out, w_patch_o, n8_o);
    cvt_f32_f16<<<(n8_o  + 255) / 256, blk, 0, stream>>>(int_w_out,   w_int_o,   n8_o);

    // ---- input projections: Q,K row-major; V transposed ----
    gemm_bias_wmma<float, _Float16, false><<<gemm_blocks(4096, 2048), blk, 0, stream>>>(
        text_tokens, w_text_in, text_b_in, text_qk, 4096, 2048, 1024, 2048);
    gemm_bias_wmma<float, _Float16, true><<<gemm_blocks(4096, 1024), blk, 0, stream>>>(
        text_tokens, w_text_in + (size_t)2048 * 1024, text_b_in + 2048,
        text_vT, 4096, 1024, 1024, 4096);
    gemm_bias_wmma<float, _Float16, false><<<gemm_blocks(16384, 2048), blk, 0, stream>>>(
        patch_groups, w_patch_in, patch_b_in, patch_qk, 16384, 2048, 1024, 2048);
    gemm_bias_wmma<float, _Float16, true><<<gemm_blocks(16384, 1024), blk, 0, stream>>>(
        patch_groups, w_patch_in + (size_t)2048 * 1024, patch_b_in + 2048,
        patch_vT, 16384, 1024, 1024, 16384);
    gemm_bias_wmma<float, _Float16, false><<<gemm_blocks(64, 1024), blk, 0, stream>>>(
        cls_tokens, w_patch_in, patch_b_in, cls_q, 64, 1024, 1024, 1024);
    gemm_bias_wmma<float, _Float16, false><<<gemm_blocks(4, 1024), blk, 0, stream>>>(
        int_token, w_int_in, int_b_in, int_q, 4, 1024, 1024, 1024);

    // ---- text causal self-attention ----
    {
        AttnParams p;
        p.q = text_qk;        p.qs = 2048;
        p.k = text_qk + 1024; p.ks = 2048;
        p.vt = text_vT;       p.vt_ld = 4096;
        p.o = text_attn;      p.os = 1024;
        p.Lq = 1024; p.Lk = 1024; p.ngroups = 4; p.causal = 1; p.clsmode = 0;
        flash_attn_wmma<<<(4 * HEADS * 64) / 8, blk, 0, stream>>>(p);
    }
    // ---- patch self-attention ----
    {
        AttnParams p;
        p.q = patch_qk;        p.qs = 2048;
        p.k = patch_qk + 1024; p.ks = 2048;
        p.vt = patch_vT;       p.vt_ld = 16384;
        p.o = patch_attn;      p.os = 1024;
        p.Lq = 256; p.Lk = 256; p.ngroups = 64; p.causal = 0; p.clsmode = 0;
        flash_attn_wmma<<<(64 * HEADS * 16) / 8, blk, 0, stream>>>(p);
    }
    // ---- CLS cross-attention ----
    {
        AttnParams p;
        p.q = cls_q;           p.qs = 1024;
        p.k = patch_qk + 1024; p.ks = 2048;
        p.vt = patch_vT;       p.vt_ld = 16384;
        p.o = cls_attn;        p.os = 1024;
        p.Lq = 1; p.Lk = 256; p.ngroups = 64; p.causal = 0; p.clsmode = 1;
        flash_attn_wmma<<<(64 * HEADS) / 8, blk, 0, stream>>>(p);
    }

    // ---- output projections (f16 A -> f32 out) ----
    gemm_bias_wmma<_Float16, float, false><<<gemm_blocks(4096, 1024), blk, 0, stream>>>(
        text_attn, w_text_o, text_b_out, out_text, 4096, 1024, 1024, 1024);
    gemm_bias_wmma<_Float16, float, false><<<gemm_blocks(16384, 1024), blk, 0, stream>>>(
        patch_attn, w_patch_o, patch_b_out, out_patch, 16384, 1024, 1024, 1024);
    gemm_bias_wmma<_Float16, float, false><<<gemm_blocks(64, 1024), blk, 0, stream>>>(
        cls_attn, w_patch_o, patch_b_out, out_cls, 64, 1024, 1024, 1024);

    // ---- INT stage ----
    {
        const int n = 4 * 1040 * (E_DIM / 8);
        gather_ctx<<<(n + 255) / 256, blk, 0, stream>>>(out_text, out_cls, ctx);
    }
    gemm_bias_wmma<_Float16, _Float16, false><<<gemm_blocks(4160, 1024), blk, 0, stream>>>(
        ctx, w_int_in + (size_t)1024 * 1024, int_b_in + 1024,
        ctx_k, 4160, 1024, 1024, 1024);
    gemm_bias_wmma<_Float16, _Float16, true><<<gemm_blocks(4160, 1024), blk, 0, stream>>>(
        ctx, w_int_in + (size_t)2048 * 1024, int_b_in + 2048,
        ctx_vT, 4160, 1024, 1024, 4160);
    {
        AttnParams p;
        p.q = int_q;     p.qs = 1024;
        p.k = ctx_k;     p.ks = 1024;
        p.vt = ctx_vT;   p.vt_ld = 4160;
        p.o = int_attn;  p.os = 1024;
        p.Lq = 1; p.Lk = 1040; p.ngroups = 4; p.causal = 0; p.clsmode = 0;
        flash_attn_wmma<<<(4 * HEADS) / 8, blk, 0, stream>>>(p);
    }
    gemm_bias_wmma<_Float16, float, false><<<gemm_blocks(4, 1024), blk, 0, stream>>>(
        int_attn, w_int_o, int_b_out, out_int, 4, 1024, 1024, 1024);
}
